// RNN_LSTM_embed_twin_58248346468906
// MI455X (gfx1250) — matrix operand order repaired and checked by reading, not compiled, not run on hardware
//
#include <hip/hip_runtime.h>
#include <hip/hip_bf16.h>

// ---------------------------------------------------------------------------
// CDNA5 (gfx1250) LSTM LM forward.
//  - bf16 WMMA (v_wmma_f32_16x16x32_bf16) for all four GEMMs
//  - wave tile 32x64 (2x4 WMMA tiles) in the GEMMs for L2-side reuse
//  - persistent LSTM scan kernel with a device-scope epoch barrier
// ---------------------------------------------------------------------------

typedef __attribute__((ext_vector_type(16))) __bf16 v16bf;
typedef __attribute__((ext_vector_type(8)))  float  v8f;

union Frag16 {
    v16bf v;
    uint4 q[2];
};

// Dimensions (fixed by the reference)
#define BB 32
#define TT 256
#define EE 256
#define HH 512
#define VV 32000
#define G4 (4 * HH)          // 2048
#define MT (BB * TT)         // 8192 tokens

// Workspace layout (byte offsets; every size is a multiple of 256)
#define OFF_SYNC 0ull                         // 2 x u32 (barrier cnt, epoch flag)
#define OFF_H    256ull                       // 2 * 32*512 bf16 ping-pong h
#define OFF_XG   (OFF_H   + 2ull*BB*HH*2ull)          // 8192*2048 f32
#define OFF_HS   (OFF_XG  + (size_t)MT*G4*4ull)       // 8192*512 bf16
#define OFF_EMB  (OFF_HS  + (size_t)MT*HH*2ull)       // 32000*256 bf16
#define OFF_WIH  (OFF_EMB + (size_t)VV*EE*2ull)       // 2048*256 bf16
#define OFF_WHH  (OFF_WIH + (size_t)G4*EE*2ull)       // 2048*512 bf16
#define OFF_FCW  (OFF_WHH + (size_t)G4*HH*2ull)       // 32000*512 bf16
#define OFF_LNW  (OFF_FCW + (size_t)VV*HH*2ull)       // 512*512 bf16

__device__ __forceinline__ unsigned short f2bf(float f) {
    union { float f; unsigned int u; } v; v.f = f;
    unsigned int u = v.u;
    // round-to-nearest-even
    unsigned int r = (u + 0x7FFFu + ((u >> 16) & 1u)) >> 16;
    return (unsigned short)r;
}

__device__ __forceinline__ float sigf(float x) {
    return 1.0f / (1.0f + __expf(-x));
}

// ---------------------------------------------------------------------------
// fp32 -> bf16 bulk conversion
// ---------------------------------------------------------------------------
__global__ void cvt_bf16_kernel(const float* __restrict__ s,
                                unsigned short* __restrict__ d, int n) {
    int i = blockIdx.x * blockDim.x + threadIdx.x;
    if (i < n) d[i] = f2bf(s[i]);
}

// ---------------------------------------------------------------------------
// zero h ping-pong + barrier state
// ---------------------------------------------------------------------------
__global__ void init_state_kernel(unsigned short* __restrict__ h2,
                                  unsigned int* __restrict__ sync) {
    int i = blockIdx.x * blockDim.x + threadIdx.x;
    if (i < 2 * BB * HH) h2[i] = 0;
    if (i < 2) sync[i] = 0u;
}

// ---------------------------------------------------------------------------
// Generic bf16 GEMM:  C[m,n] = sum_k A[m,k]*W[n,k] + bias[n]
// A is [M,K] bf16 row-major (optionally gathered through idx), W is [N,K]
// bf16 row-major.  WG = 256 threads = 8 waves (2 M x 4 N); wave tile 32x64
// (2x4 WMMA 16x16x32 bf16 tiles, 8 WMMAs per K-step); WG tile 64(M) x 256(N).
// Grids divide exactly: no bounds checks.
// A-fragment as loaded here (same k order as B): lane L elem i -> M=L%16, K=i + 16*(L/16)
// B-fragment as loaded here (same k order as A): lane L elem i -> N=L%16, K=i + 16*(L/16)
// ---------------------------------------------------------------------------
__global__ __launch_bounds__(256) void gemm_bf16_kernel(
    const unsigned short* __restrict__ A,
    const int* __restrict__ idx,              // nullable gather indices [M]
    const unsigned short* __restrict__ W,
    const float* __restrict__ bias,           // nullable [N]
    float* __restrict__ C,
    int K, int ldc)
{
    const int lane = threadIdx.x & 31;
    const int wave = threadIdx.x >> 5;
    const int wm = wave & 1;                  // 2 M-positions
    const int wn = wave >> 1;                 // 4 N-positions
    const int m0 = blockIdx.x * 64 + wm * 32;
    const int n0 = blockIdx.y * 256 + wn * 64;
    const int lmod = lane & 15;
    const int lhalf = lane >> 4;

    const unsigned short* arow[2];
    #pragma unroll
    for (int mt = 0; mt < 2; ++mt) {
        int m = m0 + mt * 16 + lmod;
        arow[mt] = idx ? (A + (size_t)idx[m] * K) : (A + (size_t)m * K);
    }
    const unsigned short* brow[4];
    #pragma unroll
    for (int nt = 0; nt < 4; ++nt) {
        int n = n0 + nt * 16 + lmod;
        brow[nt] = W + (size_t)n * K;
    }

    v8f acc[2][4] = {};
    for (int k = 0; k < K; k += 32) {
        const int ka = k + lhalf * 16;
        const int kb = k + lhalf * 16;
        v16bf a[2], b[4];
        #pragma unroll
        for (int mt = 0; mt < 2; ++mt) {
            Frag16 f;
            f.q[0] = *(const uint4*)(arow[mt] + ka);
            f.q[1] = *(const uint4*)(arow[mt] + ka + 8);
            a[mt] = f.v;
        }
        #pragma unroll
        for (int nt = 0; nt < 4; ++nt) {
            Frag16 f;
            const uint4* p = (const uint4*)(brow[nt] + kb);
            f.q[0] = p[0];
            f.q[1] = p[1];
            b[nt] = f.v;
        }
        #pragma unroll
        for (int mt = 0; mt < 2; ++mt)
            #pragma unroll
            for (int nt = 0; nt < 4; ++nt)
                acc[mt][nt] = __builtin_amdgcn_wmma_f32_16x16x32_bf16(
                    false, a[mt], false, b[nt], (short)0, acc[mt][nt],
                    false, false);
    }

    // D layout: lane L reg r -> M = r + 8*(L/16), N = L%16
    #pragma unroll
    for (int mt = 0; mt < 2; ++mt)
        #pragma unroll
        for (int nt = 0; nt < 4; ++nt) {
            int col = n0 + nt * 16 + lmod;
            float bv = bias ? bias[col] : 0.0f;
            #pragma unroll
            for (int r = 0; r < 8; ++r) {
                int row = m0 + mt * 16 + r + 8 * lhalf;
                C[(size_t)row * ldc + col] = acc[mt][nt][r] + bv;
            }
        }
}

// ---------------------------------------------------------------------------
// Persistent LSTM scan.  32 WGs x 256 threads.  WG wg owns hidden columns
// j in [wg*16, wg*16+16).  Per step: gates[32, 4x16] = h @ w_hh^T (WMMA,
// one 16x16 tile per wave), + xg_t + b_hh, activations, c in LDS, h
// ping-pong bf16 in global, hs bf16 streamed out.  Device-scope epoch
// barrier between steps.
// ---------------------------------------------------------------------------
#define LSTM_WGS 32

__global__ __launch_bounds__(256) void lstm_scan_kernel(
    const float* __restrict__ xg,             // [B*T, 2048]
    const unsigned short* __restrict__ whh,   // [2048, 512] bf16
    const float* __restrict__ b_hh,           // [2048]
    unsigned short* h2,                       // [2, 32, 512] bf16 ping-pong
    unsigned short* __restrict__ hs,          // [B*T, 512] bf16
    unsigned int* sync)                       // [0]=cnt, [1]=epoch flag
{
    __shared__ float gl[4][BB][17];           // gate tiles (+pad)
    __shared__ float cl[BB][16];              // cell state for owned columns

    const int tid  = threadIdx.x;
    const int lane = tid & 31;
    const int wave = tid >> 5;
    const int wg   = blockIdx.x;
    const int q    = wave >> 1;               // gate block 0..3 (i,f,g,o)
    const int mt   = wave & 1;                // batch half
    const int lmod = lane & 15;
    const int lhalf = lane >> 4;

    for (int p = tid; p < BB * 16; p += 256) cl[p >> 4][p & 15] = 0.0f;
    __syncthreads();

    const unsigned short* wrow =
        whh + (size_t)(q * HH + wg * 16 + lmod) * HH;   // B-frag row
    const int arowi = mt * 16 + lmod;                   // A-frag batch row

    for (int t = 0; t < TT; ++t) {
        const unsigned short* hr = h2 + (size_t)(t & 1) * BB * HH;
        unsigned short*       hw = h2 + (size_t)((t + 1) & 1) * BB * HH;

        v8f acc = {};
        for (int k = 0; k < HH; k += 32) {
            Frag16 fa, fb;
            const int ka = k + lhalf * 16;
            fa.q[0] = *(const uint4*)(hr + (size_t)arowi * HH + ka);
            fa.q[1] = *(const uint4*)(hr + (size_t)arowi * HH + ka + 8);
            const int kb = k + lhalf * 16;
            const uint4* pb = (const uint4*)(wrow + kb);
            fb.q[0] = pb[0];
            fb.q[1] = pb[1];
            acc = __builtin_amdgcn_wmma_f32_16x16x32_bf16(
                false, fa.v, false, fb.v, (short)0, acc, false, false);
        }
        #pragma unroll
        for (int r = 0; r < 8; ++r) {
            int b = mt * 16 + r + 8 * lhalf;
            gl[q][b][lmod] = acc[r];
        }
        __syncthreads();

        for (int p = tid; p < BB * 16; p += 256) {
            int b  = p >> 4;
            int j2 = p & 15;
            int j  = wg * 16 + j2;
            const float* xgb = xg + ((size_t)b * TT + t) * G4;
            float iv = gl[0][b][j2] + xgb[j]            + b_hh[j];
            float fv = gl[1][b][j2] + xgb[HH + j]       + b_hh[HH + j];
            float gv = gl[2][b][j2] + xgb[2 * HH + j]   + b_hh[2 * HH + j];
            float ov = gl[3][b][j2] + xgb[3 * HH + j]   + b_hh[3 * HH + j];
            float c  = sigf(fv) * cl[b][j2] + sigf(iv) * tanhf(gv);
            float hv = sigf(ov) * tanhf(c);
            cl[b][j2] = c;
            unsigned short hb = f2bf(hv);
            hw[(size_t)b * HH + j] = hb;
            hs[((size_t)b * TT + t) * HH + j] = hb;
        }
        __threadfence();
        __syncthreads();

        // device-scope epoch barrier
        if (tid == 0) {
            unsigned old = __hip_atomic_fetch_add(&sync[0], 1u,
                              __ATOMIC_ACQ_REL, __HIP_MEMORY_SCOPE_AGENT);
            if (old == LSTM_WGS - 1) {
                __hip_atomic_store(&sync[0], 0u,
                                   __ATOMIC_RELAXED, __HIP_MEMORY_SCOPE_AGENT);
                __hip_atomic_store(&sync[1], (unsigned)(t + 1),
                                   __ATOMIC_RELEASE, __HIP_MEMORY_SCOPE_AGENT);
            } else {
                while (__hip_atomic_load(&sync[1], __ATOMIC_ACQUIRE,
                                         __HIP_MEMORY_SCOPE_AGENT)
                       < (unsigned)(t + 1))
                    __builtin_amdgcn_s_sleep(1);
            }
        }
        __syncthreads();
        __threadfence();   // acquire visibility of other WGs' h writes
    }
}

// ---------------------------------------------------------------------------
// Host launcher
// ---------------------------------------------------------------------------
extern "C" void kernel_launch(void* const* d_in, const int* in_sizes, int n_in,
                              void* d_out, int out_size, void* d_ws, size_t ws_size,
                              hipStream_t stream) {
    const int*   x       = (const int*)  d_in[0];
    const float* embed_w = (const float*)d_in[1];
    const float* w_ih    = (const float*)d_in[2];
    const float* w_hh    = (const float*)d_in[3];
    const float* b_ih    = (const float*)d_in[4];
    const float* b_hh    = (const float*)d_in[5];
    const float* fc_w    = (const float*)d_in[6];
    const float* fc_b    = (const float*)d_in[7];
    const float* ln_w    = (const float*)d_in[8];
    const float* ln_b    = (const float*)d_in[9];

    char* ws = (char*)d_ws;
    unsigned int*   syncp = (unsigned int*)  (ws + OFF_SYNC);
    unsigned short* h2    = (unsigned short*)(ws + OFF_H);
    float*          xg    = (float*)         (ws + OFF_XG);
    unsigned short* hsb   = (unsigned short*)(ws + OFF_HS);
    unsigned short* embb  = (unsigned short*)(ws + OFF_EMB);
    unsigned short* wihb  = (unsigned short*)(ws + OFF_WIH);
    unsigned short* whhb  = (unsigned short*)(ws + OFF_WHH);
    unsigned short* fcwb  = (unsigned short*)(ws + OFF_FCW);
    unsigned short* lnwb  = (unsigned short*)(ws + OFF_LNW);

    float* out_logits = (float*)d_out;                       // [B,T,V]
    float* out_states = out_logits + (size_t)MT * VV;        // [B,T,H]

    // 1) weight conversions to bf16
    {
        int n;
        n = VV * EE; cvt_bf16_kernel<<<(n + 255) / 256, 256, 0, stream>>>(embed_w, embb, n);
        n = G4 * EE; cvt_bf16_kernel<<<(n + 255) / 256, 256, 0, stream>>>(w_ih, wihb, n);
        n = G4 * HH; cvt_bf16_kernel<<<(n + 255) / 256, 256, 0, stream>>>(w_hh, whhb, n);
        n = VV * HH; cvt_bf16_kernel<<<(n + 255) / 256, 256, 0, stream>>>(fc_w, fcwb, n);
        n = HH * HH; cvt_bf16_kernel<<<(n + 255) / 256, 256, 0, stream>>>(ln_w, lnwb, n);
    }

    // 2) zero h ping-pong + barrier state
    init_state_kernel<<<(2 * BB * HH + 255) / 256, 256, 0, stream>>>(h2, syncp);

    // 3) xg = embed[x] @ w_ih^T + b_ih   (M=8192, N=2048, K=256)
    gemm_bf16_kernel<<<dim3(MT / 64, G4 / 256), 256, 0, stream>>>(
        embb, x, wihb, b_ih, xg, EE, G4);

    // 4) LSTM scan over T=256 (persistent, 32 WGs)
    lstm_scan_kernel<<<LSTM_WGS, 256, 0, stream>>>(xg, whhb, b_hh, h2, hsb, syncp);

    // 5) logits = hs @ fc_w^T + fc_b   (M=8192, N=32000, K=512)
    gemm_bf16_kernel<<<dim3(MT / 64, VV / 256), 256, 0, stream>>>(
        hsb, nullptr, fcwb, fc_b, out_logits, HH, VV);

    // 6) states = hs @ ln_w^T + ln_b   (M=8192, N=512, K=512)
    gemm_bf16_kernel<<<dim3(MT / 64, HH / 256), 256, 0, stream>>>(
        hsb, nullptr, lnwb, ln_b, out_states, HH, HH);

    (void)in_sizes; (void)n_in; (void)out_size; (void)ws_size;
}
